// MultiHeadAttention_223338300120
// MI455X (gfx1250) — matrix operand order repaired and checked by reading, not compiled, not run on hardware
//
#include <hip/hip_runtime.h>
#include <stdint.h>

// ---------------- problem constants ----------------
#define N_HEADS   8
#define INPUT_DIM 512
#define EMBED_DIM 512
#define KEY_DIM   64
#define B_SZ      4
#define NQ        2048
#define G_SZ      2048
#define ROWS      (B_SZ * NQ)      // 8192 rows (batch folded)
#define NORM      0.125f           // 1/sqrt(64)

typedef __attribute__((ext_vector_type(16))) __bf16 v16bf;
typedef __attribute__((ext_vector_type(8)))  __bf16 v8bf;
typedef __attribute__((ext_vector_type(8)))  float  v8f;
typedef __attribute__((ext_vector_type(4)))  float  f4;

#define WMMA_BF16(a, b, c) \
  __builtin_amdgcn_wmma_f32_16x16x32_bf16(false, (a), false, (b), (short)0, (c), false, false)

// ---------------- WMMA fragment loaders (ISA 7.12.2, wave32) ----------------
// A (16x32, 16-bit): lane L holds row M=L&15; element t -> K = 8g+t (t<8),
// 16+8g+(t-8) (t>=8), g=L>>4.  => two contiguous 8-element runs.
template <typename T>
__device__ __forceinline__ v16bf load_A_bf16(const T* p, int ld, int lane) {
  const int g = (lane >> 4) & 1, m = lane & 15;
  const T* r = p + (size_t)m * ld + 8 * g;
  v8bf lo = *(const v8bf*)(r);
  v8bf hi = *(const v8bf*)(r + 16);
  return __builtin_shufflevector(lo, hi, 0, 1, 2, 3, 4, 5, 6, 7,
                                 8, 9, 10, 11, 12, 13, 14, 15);
}

// Same A fragment from an f32 source (vectorized float4 loads + convert).
__device__ __forceinline__ v16bf load_A_f32(const float* p, int ld, int lane) {
  const int g = (lane >> 4) & 1, m = lane & 15;
  const float* r = p + (size_t)m * ld + 8 * g;
  f4 x0 = *(const f4*)(r);
  f4 x1 = *(const f4*)(r + 4);
  f4 x2 = *(const f4*)(r + 16);
  f4 x3 = *(const f4*)(r + 20);
  v16bf a;
#pragma unroll
  for (int i = 0; i < 4; ++i) {
    a[i]      = (__bf16)x0[i];
    a[4 + i]  = (__bf16)x1[i];
    a[8 + i]  = (__bf16)x2[i];
    a[12 + i] = (__bf16)x3[i];
  }
  return a;
}

// B (32x16, 16-bit): lane L holds col N=L&15; element t -> K = 8g+t (t<8),
// 16+8g+(t-8) (t>=8), g=L>>4.
// Source stored K-contiguous per column: element(k,n) = p[n*ld + k]
// => two contiguous 8-element runs per lane (two b128 loads).
template <typename T>
__device__ __forceinline__ v16bf load_Bt(const T* p, int ld, int lane) {
  const int g = (lane >> 4) & 1, n = lane & 15;
  const T* r = p + (size_t)n * ld + 8 * g;
  v8bf lo = *(const v8bf*)(r);
  v8bf hi = *(const v8bf*)(r + 16);
  return __builtin_shufflevector(lo, hi, 0, 1, 2, 3, 4, 5, 6, 7,
                                 8, 9, 10, 11, 12, 13, 14, 15);
}

// ---------------- async global->LDS staging (gfx1250, ASYNCcnt) ----------------
// Copies one 4 KB tile (32 keys x 64 bf16) with 8 async instructions,
// each lane moving 16 bytes.
__device__ __forceinline__ void async_fill_4k(const __bf16* src, unsigned lds_off,
                                              int lane) {
#pragma unroll
  for (int r = 0; r < 8; ++r) {
    unsigned dst = lds_off + r * 512 + lane * 16;
    uint64_t ga = (uint64_t)(uintptr_t)src + (uint64_t)(r * 512 + lane * 16);
    asm volatile("global_load_async_to_lds_b128 %0, %1, off"
                 :: "v"(dst), "v"(ga) : "memory");
  }
}

// ---------------- kernel 0: weight prep (f32 -> bf16, K-contiguous layouts) ---
__global__ __launch_bounds__(256) void prep_weights_kernel(
    const float* __restrict__ Wq, const float* __restrict__ Wk,
    const float* __restrict__ Wv, const float* __restrict__ Wo,
    __bf16* __restrict__ WqT, __bf16* __restrict__ WkT,
    __bf16* __restrict__ WvT, __bf16* __restrict__ WoT) {
  const int tid = blockIdx.x * blockDim.x + threadIdx.x;
  const int PER = N_HEADS * INPUT_DIM * KEY_DIM;  // 262144
  if (tid < 3 * PER) {
    const int m = tid / PER, r = tid % PER;
    const int hd = r / (INPUT_DIM * KEY_DIM);
    const int rem = r % (INPUT_DIM * KEY_DIM);
    const int k = rem / KEY_DIM, n = rem % KEY_DIM;
    const float* W = (m == 0) ? Wq : (m == 1 ? Wk : Wv);
    __bf16*      T = (m == 0) ? WqT : (m == 1 ? WkT : WvT);
    T[(size_t)hd * KEY_DIM * INPUT_DIM + (size_t)n * INPUT_DIM + k] = (__bf16)W[r];
  } else {
    const int r = tid - 3 * PER;
    if (r < EMBED_DIM * EMBED_DIM) {
      const int k = r / EMBED_DIM, n = r % EMBED_DIM;
      WoT[(size_t)n * EMBED_DIM + k] = (__bf16)Wo[r];
    }
  }
}

// ---------------- kernel 1: fused QKV projection ----------------
// wave tile id bits: [ct:2][rt:9][hd:3][pj:2]; 3*8*512*4 = 49152 wave tiles.
// Q,K stored [h][row][64]; V stored transposed [h][dim][row] for P*V B-frags.
__global__ __launch_bounds__(256) void qkv_proj_kernel(
    const float* __restrict__ q, const float* __restrict__ h,
    const __bf16* __restrict__ WqT, const __bf16* __restrict__ WkT,
    const __bf16* __restrict__ WvT,
    __bf16* __restrict__ Qb, __bf16* __restrict__ Kb, __bf16* __restrict__ Vt) {
  const int wave = blockIdx.x * (blockDim.x >> 5) + (threadIdx.x >> 5);
  const int lane = threadIdx.x & 31;
  const int ct = wave & 3;
  const int rt = (wave >> 2) & 511;
  const int hd = (wave >> 11) & 7;
  const int pj = wave >> 14;             // wave-uniform
  if (pj > 2) return;

  const float*  X  = (pj == 0) ? q : h;
  const __bf16* WT = (pj == 0) ? WqT : (pj == 1 ? WkT : WvT);

  const int r0 = rt * 16, n0 = ct * 16;
  v8f acc = {};
#pragma unroll 4
  for (int kc = 0; kc < INPUT_DIM; kc += 32) {
    v16bf a = load_A_f32(X + (size_t)r0 * INPUT_DIM + kc, INPUT_DIM, lane);
    v16bf b = load_Bt(WT + (size_t)hd * KEY_DIM * INPUT_DIM +
                          (size_t)n0 * INPUT_DIM + kc, INPUT_DIM, lane);
    acc = WMMA_BF16(a, b, acc);
  }
  const int g = lane >> 4, n = lane & 15;
  if (pj < 2) {
    __bf16* Dst = (pj == 0) ? Qb : Kb;
#pragma unroll
    for (int i = 0; i < 8; ++i) {
      const int row = r0 + g * 8 + i;
      Dst[((size_t)hd * ROWS + row) * KEY_DIM + n0 + n] = (__bf16)acc[i];
    }
  } else {
    v8bf vv;                              // transposed: rows are contiguous
#pragma unroll
    for (int i = 0; i < 8; ++i) vv[i] = (__bf16)acc[i];
    *(v8bf*)(Vt + ((size_t)hd * KEY_DIM + n0 + n) * ROWS + r0 + 8 * g) = vv;
  }
}

// ---------------- kernel 2: fused masked attention (flash style) ----------------
// One wave owns a 16-row query tile of one (head, batch). 8 waves/block.
// K tiles are double-buffered in LDS via global_load_async_to_lds_b128.
__global__ __launch_bounds__(256) void attn_kernel(
    const __bf16* __restrict__ Qb, const __bf16* __restrict__ Kb,
    const __bf16* __restrict__ Vt, const unsigned char* __restrict__ mask,
    __bf16* __restrict__ Hd) {
  __shared__ __align__(16) __bf16 lds_k[8][2][32 * KEY_DIM];  // 64 KB: K tiles
  __shared__ __align__(16) __bf16 lds_p[8][16 * 32];          // 8 KB: P relayout

  const int wslot = threadIdx.x >> 5;
  const int wave  = blockIdx.x * (blockDim.x >> 5) + wslot;
  const int lane  = threadIdx.x & 31;
  const int qt = wave & 127;
  const int b  = (wave >> 7) & 3;
  const int hd = wave >> 9;
  if (hd >= N_HEADS) return;             // wave-uniform

  const int q0 = qt * 16;
  const __bf16* Qp = Qb + ((size_t)hd * ROWS + b * NQ + q0) * KEY_DIM;
  const __bf16* Kp = Kb + ((size_t)hd * ROWS + b * G_SZ) * KEY_DIM;
  const __bf16* Vp = Vt + (size_t)hd * KEY_DIM * ROWS + b * G_SZ;  // [dim][key]
  const unsigned char* mp = mask + ((size_t)b * NQ + q0) * G_SZ;

  const v16bf aq0 = load_A_bf16(Qp, KEY_DIM, lane);        // K-chunk 0..31
  const v16bf aq1 = load_A_bf16(Qp + 32, KEY_DIM, lane);   // K-chunk 32..63

  v8f acc[4] = {v8f{}, v8f{}, v8f{}, v8f{}};   // O: 16 x 64
  float mrow[8], lrow[8];
#pragma unroll
  for (int i = 0; i < 8; ++i) { mrow[i] = -__builtin_inff(); lrow[i] = 0.0f; }

  const int g = lane >> 4, n = lane & 15;
  __bf16* pt = lds_p[wslot];
  const unsigned kbuf0 = (unsigned)(uintptr_t)&lds_k[wslot][0][0];  // low 32b = LDS offset
  const unsigned kbuf1 = (unsigned)(uintptr_t)&lds_k[wslot][1][0];

  async_fill_4k(Kp, kbuf0, lane);        // prologue: stage tile 0

  for (int g0 = 0; g0 < G_SZ; g0 += 32) {
    const int cur = (g0 >> 5) & 1;
    if (g0 + 32 < G_SZ) {                // stage next tile, then wait for current
      async_fill_4k(Kp + (size_t)(g0 + 32) * KEY_DIM, cur ? kbuf0 : kbuf1, lane);
      asm volatile("s_wait_asynccnt 0x8" ::: "memory");
      __builtin_prefetch(Vp + (size_t)(g0 + 32), 0, 3);    // warm next V rows
    } else {
      asm volatile("s_wait_asynccnt 0x0" ::: "memory");
    }
    const __bf16* ldsK = &lds_k[wslot][cur][0];            // [32 keys][64 dims]

    // ---- scores S = Q K^T for keys [g0, g0+32) ----
    v8f s0 = {}, s1 = {};
    s0 = WMMA_BF16(aq0, load_Bt(ldsK, KEY_DIM, lane), s0);
    s0 = WMMA_BF16(aq1, load_Bt(ldsK + 32, KEY_DIM, lane), s0);
    s1 = WMMA_BF16(aq0, load_Bt(ldsK + 16 * KEY_DIM, KEY_DIM, lane), s1);
    s1 = WMMA_BF16(aq1, load_Bt(ldsK + 16 * KEY_DIM + 32, KEY_DIM, lane), s1);

    // ---- masked online softmax (f32), per C-layout row i+8g ----
    float rescale[8];
#pragma unroll
    for (int i = 0; i < 8; ++i) {
      const int qrow = g * 8 + i;
      const bool mk0 = mp[(size_t)qrow * G_SZ + g0 + n] != 0;
      const bool mk1 = mp[(size_t)qrow * G_SZ + g0 + 16 + n] != 0;
      const float v0 = mk0 ? -__builtin_inff() : s0[i] * NORM;
      const float v1 = mk1 ? -__builtin_inff() : s1[i] * NORM;
      float rm = fmaxf(v0, v1);           // 16-lane row reductions
      rm = fmaxf(rm, __shfl_xor(rm, 1));
      rm = fmaxf(rm, __shfl_xor(rm, 2));
      rm = fmaxf(rm, __shfl_xor(rm, 4));
      rm = fmaxf(rm, __shfl_xor(rm, 8));
      const float mnew = fmaxf(mrow[i], rm);
      const float p0 = mk0 ? 0.0f : __expf(v0 - mnew);
      const float p1 = mk1 ? 0.0f : __expf(v1 - mnew);
      const float corr = (mnew == mrow[i]) ? 1.0f : __expf(mrow[i] - mnew);
      float rs = p0 + p1;
      rs += __shfl_xor(rs, 1);
      rs += __shfl_xor(rs, 2);
      rs += __shfl_xor(rs, 4);
      rs += __shfl_xor(rs, 8);
      lrow[i] = lrow[i] * corr + rs;
      mrow[i] = mnew;
      rescale[i] = corr;
      pt[(size_t)(g * 8 + i) * 32 + n]      = (__bf16)p0;   // P tile 16x32
      pt[(size_t)(g * 8 + i) * 32 + 16 + n] = (__bf16)p1;
    }
#pragma unroll
    for (int c = 0; c < 4; ++c)
#pragma unroll
      for (int i = 0; i < 8; ++i) acc[c][i] *= rescale[i];

    asm volatile("" ::: "memory");       // DS is in-order within a wave
    const v16bf ap = load_A_bf16(pt, 32, lane);
#pragma unroll
    for (int c = 0; c < 4; ++c) {
      // V^T is [dim][key] -> B frag is two 8-key runs per lane
      v16bf bv = load_Bt(Vp + ((size_t)c * 16) * ROWS + g0, ROWS, lane);
      acc[c] = WMMA_BF16(ap, bv, acc[c]);
    }
  }

  // ---- finalize: divide by l (all-masked rows -> 0, matching reference) ----
  float inv[8];
#pragma unroll
  for (int i = 0; i < 8; ++i) inv[i] = (lrow[i] > 0.0f) ? (1.0f / lrow[i]) : 0.0f;
#pragma unroll
  for (int i = 0; i < 8; ++i) {
    const int row = q0 + g * 8 + i;
#pragma unroll
    for (int c = 0; c < 4; ++c)
      Hd[((size_t)b * NQ + row) * EMBED_DIM + hd * KEY_DIM + c * 16 + n] =
          (__bf16)(acc[c][i] * inv[i]);
  }
}

// ---------------- kernel 3: output projection ----------------
__global__ __launch_bounds__(256) void outproj_kernel(
    const __bf16* __restrict__ Hd, const __bf16* __restrict__ WoT,
    float* __restrict__ out) {
  const int wave = blockIdx.x * (blockDim.x >> 5) + (threadIdx.x >> 5);
  const int lane = threadIdx.x & 31;
  const int ct = wave & 31;
  const int rt = wave >> 5;
  if (rt >= ROWS / 16) return;           // wave-uniform

  const int r0 = rt * 16, n0 = ct * 16;
  v8f acc = {};
#pragma unroll 4
  for (int kc = 0; kc < EMBED_DIM; kc += 32) {
    v16bf a = load_A_bf16(Hd + (size_t)r0 * EMBED_DIM + kc, EMBED_DIM, lane);
    v16bf b = load_Bt(WoT + (size_t)n0 * EMBED_DIM + kc, EMBED_DIM, lane);
    acc = WMMA_BF16(a, b, acc);
  }
  const int g = lane >> 4, n = lane & 15;
#pragma unroll
  for (int i = 0; i < 8; ++i)
    out[(size_t)(r0 + g * 8 + i) * EMBED_DIM + n0 + n] = acc[i];
}

// ---------------- host-side launch ----------------
extern "C" void kernel_launch(void* const* d_in, const int* in_sizes, int n_in,
                              void* d_out, int out_size, void* d_ws, size_t ws_size,
                              hipStream_t stream) {
  const float* q  = (const float*)d_in[0];
  const float* h  = (const float*)d_in[1];
  const unsigned char* mask = (const unsigned char*)d_in[2];  // jax bool = 1 byte
  const float* Wq = (const float*)d_in[3];
  const float* Wk = (const float*)d_in[4];
  const float* Wv = (const float*)d_in[5];
  const float* Wo = (const float*)d_in[6];
  float* out = (float*)d_out;

  // workspace: Qb | Kb | Vt | Hd (8 MB each, bf16) | WqT WkT WvT WoT (512 KB each)
  char* ws = (char*)d_ws;
  __bf16* Qb  = (__bf16*)(ws);
  __bf16* Kb  = (__bf16*)(ws + ((size_t)8 << 20));
  __bf16* Vt  = (__bf16*)(ws + ((size_t)16 << 20));
  __bf16* Hd  = (__bf16*)(ws + ((size_t)24 << 20));
  __bf16* WqT = (__bf16*)(ws + ((size_t)32 << 20));
  __bf16* WkT = (__bf16*)(ws + ((size_t)32 << 20) + ((size_t)512 << 10));
  __bf16* WvT = (__bf16*)(ws + ((size_t)32 << 20) + ((size_t)1024 << 10));
  __bf16* WoT = (__bf16*)(ws + ((size_t)32 << 20) + ((size_t)1536 << 10));

  prep_weights_kernel<<<4096, 256, 0, stream>>>(Wq, Wk, Wv, Wo, WqT, WkT, WvT, WoT);
  qkv_proj_kernel<<<6144, 256, 0, stream>>>(q, h, WqT, WkT, WvT, Qb, Kb, Vt);
  attn_kernel<<<512, 256, 0, stream>>>(Qb, Kb, Vt, mask, Hd);
  outproj_kernel<<<2048, 256, 0, stream>>>(Hd, WoT, out);
}
